// GINEdge_83408264888790
// MI455X (gfx1250) — matrix operand order repaired and checked by reading, not compiled, not run on hardware
//
#include <hip/hip_runtime.h>
#include <hip/hip_bf16.h>

// ---------------------------------------------------------------------------
// GIN edge predictor for MI455X (gfx1250, wave32, WMMA).
// Node pipeline: fp32 VALU (small 64x64 GEMMs + training-mode BN).
// Edge predictor (dominant 52 GFLOP): f16 v_wmma_f32_16x16x32_f16, with
// W1 pre-swizzled into B-fragment order so LDS staging is a coalesced copy.
// ---------------------------------------------------------------------------

typedef __attribute__((ext_vector_type(16))) _Float16 v16h;
typedef __attribute__((ext_vector_type(8)))  _Float16 v8h;
typedef __attribute__((ext_vector_type(8)))  float    v8f;

#define HID 64
#define BN_EPS 1e-5f

// ------------------------------ utility kernels ----------------------------

__global__ void zero_f32(float* __restrict__ p, long long n) {
  long long t = (long long)blockIdx.x * blockDim.x + threadIdx.x;
  if (t < n) p[t] = 0.0f;
}

__global__ void f32_to_f16(const float* __restrict__ x, _Float16* __restrict__ y,
                           long long n) {
  long long t = (long long)blockIdx.x * blockDim.x + threadIdx.x;
  if (t < n) y[t] = (_Float16)x[t];
}

// Convert pred W1 [4][128][64] fp32 -> f16 in WMMA B-fragment order:
//   frag f = ks*4+nt (ks = K/32 step, nt = N/16 tile)
//   W1h[l][(f*32+ln)*16+j] = W1[l][K][n],
//     K = (f>>2)*32 + (j<8 ? j : j+8) + (ln>=16 ? 8 : 0),  n = (f&3)*16 + (ln&15)
// (B 32x16 f16 layout: lane holds column n = ln&15; halves j -> K = (j<8?j:j+8) + 8*hi)
__global__ void prep_w1_frag(const float* __restrict__ W1, _Float16* __restrict__ W1h) {
  int t = blockIdx.x * blockDim.x + threadIdx.x;  // 0 .. 4*8192-1
  if (t >= 4 * 8192) return;
  int l   = t >> 13;
  int idx = t & 8191;
  int j  = idx & 15;
  int ln = (idx >> 4) & 31;
  int f  = idx >> 9;
  int K  = ((f >> 2) << 5) + ((j < 8) ? j : j + 8) + ((ln >= 16) ? 8 : 0);
  int n  = ((f & 3) << 4) + (ln & 15);
  W1h[t] = (_Float16)W1[l * 8192 + K * HID + n];
}

// Y[node,f] = bias[f] + sum_k (X[node,k] (+ Xadd[node,k])) * W[k,f]   (64x64 W)
__global__ __launch_bounds__(256) void gemm_node(
    const float* __restrict__ X, const float* __restrict__ Xadd,
    const float* __restrict__ W, const float* __restrict__ bias,
    float* __restrict__ Y, int Nn) {
  __shared__ float sW[HID * HID];   // 16 KB
  __shared__ float sX[16 * HID];    // 4 KB, 16 nodes per block
  const int tid = threadIdx.x;
  const int n0 = blockIdx.x * 16;
  for (int i = tid; i < HID * HID; i += 256) sW[i] = W[i];
  for (int i = tid; i < 16 * HID; i += 256) {
    int node = n0 + (i >> 6);
    float v = 0.0f;
    if (node < Nn) {
      v = X[(size_t)node * HID + (i & 63)];
      if (Xadd) v += Xadd[(size_t)node * HID + (i & 63)];
    }
    sX[i] = v;
  }
  __syncthreads();
  for (int r = 0; r < 4; ++r) {
    int idx = tid + r * 256;
    int ln = idx >> 6;
    int f  = idx & 63;
    int node = n0 + ln;
    if (node >= Nn) continue;
    float acc = bias[f];
#pragma unroll
    for (int k = 0; k < HID; ++k) acc += sX[ln * HID + k] * sW[k * HID + f];
    Y[(size_t)node * HID + f] = acc;
  }
}

// agg[dst[e], f] += hn[src[e], f]   (GIN sum aggregation)
__global__ void scatter_add(const float* __restrict__ hn, const int* __restrict__ src,
                            const int* __restrict__ dst, float* __restrict__ agg,
                            long long total) {
  long long t = (long long)blockIdx.x * blockDim.x + threadIdx.x;
  if (t >= total) return;
  int e = (int)(t >> 6);
  int f = (int)(t & 63);
  atomicAdd(&agg[(size_t)dst[e] * HID + f], hn[(size_t)src[e] * HID + f]);
}

// stats[0..63] += sum over nodes, stats[64..127] += sum of squares (per feature)
__global__ __launch_bounds__(256) void bn_stats(const float* __restrict__ X,
                                                float* __restrict__ stats, int Nn) {
  __shared__ float sS[256], sQ[256];
  const int tid = threadIdx.x;
  const int f = tid & 63, g = tid >> 6;
  float s = 0.0f, q = 0.0f;
  const int base = blockIdx.x * 256;
  for (int i = 0; i < 64; ++i) {
    int node = base + g + i * 4;
    if (node < Nn) {
      float v = X[(size_t)node * HID + f];
      s += v; q += v * v;
    }
  }
  sS[tid] = s; sQ[tid] = q;
  __syncthreads();
  if (g == 0) {
    s = sS[f] + sS[64 + f] + sS[128 + f] + sS[192 + f];
    q = sQ[f] + sQ[64 + f] + sQ[128 + f] + sQ[192 + f];
    atomicAdd(&stats[f], s);
    atomicAdd(&stats[64 + f], q);
  }
}

// Y = relu(gamma*(X-mean)*rsqrt(var+eps)+beta); optional f16 copy
__global__ void bn_relu(const float* __restrict__ X, const float* __restrict__ stats,
                        const float* __restrict__ gamma, const float* __restrict__ beta,
                        float* __restrict__ Y, _Float16* __restrict__ Yh, int Nn) {
  long long t = (long long)blockIdx.x * blockDim.x + threadIdx.x;
  long long total = (long long)Nn * HID;
  if (t >= total) return;
  int f = (int)(t & 63);
  float invN = 1.0f / (float)Nn;
  float mean = stats[f] * invN;
  float var  = stats[64 + f] * invN - mean * mean;
  float scale = rsqrtf(var + BN_EPS);
  float v = gamma[f] * (X[t] - mean) * scale + beta[f];
  v = v > 0.0f ? v : 0.0f;
  Y[t] = v;
  if (Yh) Yh[t] = (_Float16)v;
}

// ------------------------- edge predictor (WMMA) ---------------------------
// For each 16-edge tile and each of 4 layers:
//   A(16x128) = [hnh[l][src] | hnh[l][dst]]  (f16, gathered)
//   H(16x64)  = relu(A @ W1h[l] + b1[l])     via 16x v_wmma_f32_16x16x32_f16
//   score    += H @ W2[l] + b2[l]            (64->2, per-lane fp32, float4 LDS)
// Fragment layouts per CDNA5 ISA 7.12.2:
//   A 16x32 f16: lane m=lane&15; halves j: K = (j<8?j:j+8) + (lane>=16?8:0)
//   B 32x16 f16: pre-swizzled in global memory (see prep_w1_frag)
//   C/D 16x16 f32: VGPR r: M = (lane>=16?8:0)+r, N = lane&15
__global__ __launch_bounds__(256) void edge_pred_wmma(
    const _Float16* __restrict__ hnh,  // [4][N][64]
    const int* __restrict__ src, const int* __restrict__ dst,
    const _Float16* __restrict__ W1h,  // [4][8192] fragment-swizzled
    const float* __restrict__ b1,      // [4][64]
    const float* __restrict__ W2,      // [4][64][2]
    const float* __restrict__ b2,      // [4][2]
    float* __restrict__ out, int E, int N) {
  __shared__ __align__(32) _Float16 sW1[16 * 32 * 16];  // 16 B-frags, 16 KB
  __shared__ float sB1[64];
  __shared__ __align__(16) float sW2[128];              // transposed: [2][64]
  __shared__ float sB2[2];
  __shared__ __align__(16) float sH[8][16 * HID];       // per-wave hidden, 32 KB

  const int tid  = threadIdx.x;
  const int wave = tid >> 5;
  const int lane = tid & 31;
  const int m    = lane & 15;
  const int hi   = lane >> 4;                 // 0 or 1
  const int tile = blockIdx.x * 8 + wave;     // 16 edges per tile

  int e = tile * 16 + m;
  int sN = (e < E) ? src[e] : 0;
  int dN = (e < E) ? dst[e] : 0;

  const int oe   = lane >> 1;                 // output phase: edge in tile
  const int ocls = lane & 1;                  // output phase: class
  float scoreAcc = 0.0f;

  const long long n64 = (long long)N * HID;

  for (int l = 0; l < 4; ++l) {
    // -- stage this layer's weights into LDS (coalesced vector copy) --
    {
      const v8h* wsrc = (const v8h*)(W1h + (size_t)l * 8192);
      v8h* wdst = (v8h*)sW1;
      for (int i = tid; i < 1024; i += 256) wdst[i] = wsrc[i];
    }
    if (tid < 64)  sB1[tid] = b1[l * 64 + tid];
    if (tid < 128) sW2[(tid & 1) * 64 + (tid >> 1)] = W2[l * 128 + tid];
    if (tid < 2)   sB2[tid] = b2[l * 2 + tid];
    __syncthreads();

    const _Float16* base = hnh + (size_t)l * n64;

    v8f z = {0, 0, 0, 0, 0, 0, 0, 0};
    v8f acc[4] = {z, z, z, z};

#pragma unroll
    for (int ks = 0; ks < 4; ++ks) {          // K step of 32 within 128
      // A fragment: K 0..63 -> src features, 64..127 -> dst features
      const _Float16* row =
          base + (size_t)((ks < 2) ? sN : dN) * HID + (ks & 1) * 32 + hi * 8;
      union { v16h v; v8h h[2]; } A;
      A.h[0] = *(const v8h*)(row);            // K-chunk [lo, lo+8)
      A.h[1] = *(const v8h*)(row + 16);       // K-chunk [lo+16, lo+24)
#pragma unroll
      for (int nt = 0; nt < 4; ++nt) {
        const v16h* bf = (const v16h*)&sW1[((ks * 4 + nt) * 32 + lane) * 16];
        acc[nt] = __builtin_amdgcn_wmma_f32_16x16x32_f16(
            false, A.v, false, *bf, (short)0, acc[nt], false, false);
      }
    }

    // -- bias + relu, spill H to LDS --
#pragma unroll
    for (int nt = 0; nt < 4; ++nt) {
      int col = nt * 16 + m;
      float bias = sB1[col];
#pragma unroll
      for (int r = 0; r < 8; ++r) {
        int rowM = hi * 8 + r;
        float v = acc[nt][r] + bias;
        sH[wave][rowM * HID + col] = v > 0.0f ? v : 0.0f;
      }
    }
    __syncthreads();

    // -- H(16x64) @ W2(64x2) + b2, accumulate score (float4 LDS reads) --
    {
      float d = sB2[ocls];
      const float4* hp = (const float4*)&sH[wave][oe * HID];
      const float4* wp = (const float4*)&sW2[ocls * 64];
#pragma unroll
      for (int k = 0; k < 16; ++k) {
        float4 a = hp[k];
        float4 b = wp[k];
        d += a.x * b.x + a.y * b.y + a.z * b.z + a.w * b.w;
      }
      scoreAcc += d;
    }
    __syncthreads();   // before next layer overwrites sW1 / sH
  }

  int eo = tile * 16 + oe;
  if (eo < E) out[eo * 2 + ocls] = scoreAcc;
}

// ------------------------------- launcher ----------------------------------

extern "C" void kernel_launch(void* const* d_in, const int* in_sizes, int n_in,
                              void* d_out, int out_size, void* d_ws, size_t ws_size,
                              hipStream_t stream) {
  const float* h    = (const float*)d_in[0];
  const int*   src  = (const int*)d_in[1];
  const int*   dst  = (const int*)d_in[2];
  const float* embW = (const float*)d_in[3];
  const float* embB = (const float*)d_in[4];
  const float* mW1  = (const float*)d_in[5];
  const float* mB1  = (const float*)d_in[6];
  const float* mBNg = (const float*)d_in[7];
  const float* mBNb = (const float*)d_in[8];
  const float* mW2  = (const float*)d_in[9];
  const float* mB2  = (const float*)d_in[10];
  const float* aBNg = (const float*)d_in[11];
  const float* aBNb = (const float*)d_in[12];
  const float* oBNg = (const float*)d_in[13];
  const float* oBNb = (const float*)d_in[14];
  const float* pW1  = (const float*)d_in[15];
  const float* pB1  = (const float*)d_in[16];
  const float* pW2  = (const float*)d_in[17];
  const float* pB2  = (const float*)d_in[18];

  const int N = in_sizes[0] / HID;
  const int E = in_sizes[1];
  const long long N64 = (long long)N * HID;

  // workspace carve-up (all chunks 256B aligned)
  char* w = (char*)d_ws;
  float* hn    = (float*)w;  w += N64 * 4;
  float* agg   = (float*)w;  w += N64 * 4;
  float* t1    = (float*)w;  w += N64 * 4;
  float* t2    = (float*)w;  w += N64 * 4;
  float* stats = (float*)w;  w += 256 * 4;            // 128 used
  _Float16* w1h = (_Float16*)w;  w += 4 * 8192 * 2;   // fragment-swizzled
  _Float16* hnh = (_Float16*)w;                        // [4][N][64] f16

  const dim3 B(256);
  const int gN16  = (N + 15) / 16;
  const int gN256 = (N + 255) / 256;
  const int gElem = (int)((N64 + 255) / 256);

  // convert + fragment-swizzle predictor W1 once per call
  prep_w1_frag<<<(4 * 8192 + 255) / 256, B, 0, stream>>>(pW1, w1h);

  // embedding: hn = h @ embW + embB ; hnh[0] = f16(hn)
  gemm_node<<<gN16, B, 0, stream>>>(h, nullptr, embW, embB, hn, N);
  f32_to_f16<<<gElem, B, 0, stream>>>(hn, hnh, N64);

  for (int l = 0; l < 3; ++l) {
    // GIN aggregation
    zero_f32<<<gElem, B, 0, stream>>>(agg, N64);
    long long EW = (long long)E * HID;
    scatter_add<<<(int)((EW + 255) / 256), B, 0, stream>>>(hn, src, dst, agg, EW);
    // t1 = (hn + agg) @ W1 + b1 ; BN ; relu
    gemm_node<<<gN16, B, 0, stream>>>(hn, agg, mW1 + l * HID * HID, mB1 + l * HID, t1, N);
    zero_f32<<<1, B, 0, stream>>>(stats, 128);
    bn_stats<<<gN256, B, 0, stream>>>(t1, stats, N);
    bn_relu<<<gElem, B, 0, stream>>>(t1, stats, mBNg + l * HID, mBNb + l * HID, t1, nullptr, N);
    // t2 = t1 @ W2 + b2 ; apply-BN ; relu ; out-BN ; relu -> hn, hnh[l+1]
    gemm_node<<<gN16, B, 0, stream>>>(t1, nullptr, mW2 + l * HID * HID, mB2 + l * HID, t2, N);
    zero_f32<<<1, B, 0, stream>>>(stats, 128);
    bn_stats<<<gN256, B, 0, stream>>>(t2, stats, N);
    bn_relu<<<gElem, B, 0, stream>>>(t2, stats, aBNg + l * HID, aBNb + l * HID, t2, nullptr, N);
    zero_f32<<<1, B, 0, stream>>>(stats, 128);
    bn_stats<<<gN256, B, 0, stream>>>(t2, stats, N);
    bn_relu<<<gElem, B, 0, stream>>>(t2, stats, oBNg + l * HID, oBNb + l * HID,
                                     hn, hnh + (long long)(l + 1) * N64, N);
  }

  // edge predictor: dominant GEMM on tensor cores
  int tiles  = (E + 15) / 16;
  int blocks = (tiles + 7) / 8;
  edge_pred_wmma<<<blocks, B, 0, stream>>>(hnh, src, dst, w1h, pB1, pW2, pB2,
                                           (float*)d_out, E, N);
}
